// QuantizedChannelMixer_78211354460753
// MI455X (gfx1250) — 8-bit matrix operand order repaired and checked by reading, compiled, not run on hardware
//
#include <hip/hip_runtime.h>
#include <hip/hip_bf16.h>
#include <stdint.h>

typedef __attribute__((ext_vector_type(8))) int v8i;

#define D_MODEL 2048
#define HIDDEN  8192
#define TOKENS  8192
#define QB      127.0f

static const size_t OFF_AQ    = 0;
static const size_t OFF_WGVQ  = 16777216;
static const size_t OFF_WOUTQ = 50331648;
static const size_t OFF_HQ    = 67108864;
static const size_t OFF_HF    = 134217728;
static const size_t OFF_INVA  = 268435456;
static const size_t OFF_INVH  = 268468224;
static const size_t OFF_PART  = 268500992;
static const size_t OFF_SCAL  = 268509184;

__device__ __forceinline__ void async_copy16(void* lds_ptr, const void* gptr) {
  uint32_t lds_addr = (uint32_t)(uintptr_t)lds_ptr;
  uint64_t gaddr    = (uint64_t)(uintptr_t)gptr;
  asm volatile("global_load_async_to_lds_b128 %0, %1, off"
               :: "v"(lds_addr), "v"(gaddr)
               : "memory");
}
__device__ __forceinline__ void wait_async() {
  asm volatile("s_wait_asynccnt 0x0" ::: "memory");
}

__global__ __launch_bounds__(256) void absmean_partial(const float* __restrict__ w,
                                                       size_t n, float* __restrict__ part) {
  __shared__ float red[256];
  const int tid = threadIdx.x;
  float s = 0.f;
  for (size_t i = (size_t)blockIdx.x * 256 + tid; i < n; i += (size_t)gridDim.x * 256)
    s += fabsf(w[i]);
  red[tid] = s; __syncthreads();
  for (int st = 128; st > 0; st >>= 1) { if (tid < st) red[tid] += red[tid + st]; __syncthreads(); }
  if (tid == 0) part[blockIdx.x] = red[0];
}

__global__ __launch_bounds__(256) void finalize_scale(const float* __restrict__ part, int np,
                                                      float inv_count, float* __restrict__ out) {
  __shared__ float red[256];
  const int tid = threadIdx.x;
  float s = 0.f;
  for (int i = tid; i < np; i += 256) s += part[i];
  red[tid] = s; __syncthreads();
  for (int st = 128; st > 0; st >>= 1) { if (tid < st) red[tid] += red[tid + st]; __syncthreads(); }
  if (tid == 0) out[0] = fmaxf(red[0] * inv_count, 1e-5f);
}

__global__ __launch_bounds__(256) void quant_weight(const float* __restrict__ w,
                                                    const float* __restrict__ sptr,
                                                    int8_t* __restrict__ q, size_t n) {
  const float inv_s = 1.f / sptr[0];
  for (size_t i = (size_t)blockIdx.x * 256 + threadIdx.x; i < n; i += (size_t)gridDim.x * 256) {
    float t = rintf(w[i] * inv_s);
    t = fminf(1.f, fmaxf(-1.f, t));
    q[i] = (int8_t)t;
  }
}

__global__ __launch_bounds__(256) void rmsnorm_actquant(const float* __restrict__ X,
                                                        const float* __restrict__ G,
                                                        int8_t* __restrict__ Q,
                                                        float* __restrict__ inv_a) {
  __shared__ float red[256];
  const int tid = threadIdx.x;
  const size_t base = (size_t)blockIdx.x * D_MODEL;
  float v[8];
  float ss = 0.f;
#pragma unroll
  for (int i = 0; i < 8; ++i) { v[i] = X[base + tid + i * 256]; ss += v[i] * v[i]; }
  red[tid] = ss; __syncthreads();
  for (int st = 128; st > 0; st >>= 1) { if (tid < st) red[tid] += red[tid + st]; __syncthreads(); }
  const float rstd = rsqrtf(red[0] * (1.f / D_MODEL) + 1e-5f);
  __syncthreads();
  float y[8]; float amax = 0.f;
#pragma unroll
  for (int i = 0; i < 8; ++i) { y[i] = v[i] * rstd * G[tid + i * 256]; amax = fmaxf(amax, fabsf(y[i])); }
  red[tid] = amax; __syncthreads();
  for (int st = 128; st > 0; st >>= 1) { if (tid < st) red[tid] = fmaxf(red[tid], red[tid + st]); __syncthreads(); }
  const float mx = fmaxf(red[0], 1e-5f);
  const float sc = QB / mx;
#pragma unroll
  for (int i = 0; i < 8; ++i) {
    float t = fminf(QB, fmaxf(-QB, rintf(y[i] * sc)));
    Q[base + tid + i * 256] = (int8_t)t;
  }
  if (tid == 0) inv_a[blockIdx.x] = mx / QB;
}

__global__ __launch_bounds__(256) void actquant_row(const __hip_bfloat16* __restrict__ H,
                                                    int8_t* __restrict__ Q,
                                                    float* __restrict__ inv_h) {
  __shared__ float red[256];
  const int tid = threadIdx.x;
  const size_t base = (size_t)blockIdx.x * HIDDEN;
  float amax = 0.f;
  for (int i = tid; i < HIDDEN; i += 256) amax = fmaxf(amax, fabsf(__bfloat162float(H[base + i])));
  red[tid] = amax; __syncthreads();
  for (int st = 128; st > 0; st >>= 1) { if (tid < st) red[tid] = fmaxf(red[tid], red[tid + st]); __syncthreads(); }
  const float mx = fmaxf(red[0], 1e-5f);
  const float sc = QB / mx;
  for (int i = tid; i < HIDDEN; i += 256) {
    float t = fminf(QB, fmaxf(-QB, rintf(__bfloat162float(H[base + i]) * sc)));
    Q[base + i] = (int8_t)t;
  }
  if (tid == 0) inv_h[blockIdx.x] = mx / QB;
}

__device__ __forceinline__ v8i load_a_frag(const int8_t* As, int row, int hi) {
  const int8_t* p = As + row * 80 + hi * 8;
  int2 p0 = *(const int2*)(p);
  int2 p1 = *(const int2*)(p + 16);
  int2 p2 = *(const int2*)(p + 32);
  int2 p3 = *(const int2*)(p + 48);
  v8i a = {p0.x, p0.y, p1.x, p1.y, p2.x, p2.y, p3.x, p3.y};
  return a;
}
__device__ __forceinline__ v8i load_b_frag(const int8_t* Bs, int col, int hi) {
  const int8_t* p = Bs + col * 80 + hi * 8;
  int2 q0 = *(const int2*)(p);
  int2 q1 = *(const int2*)(p + 16);
  int2 q2 = *(const int2*)(p + 32);
  int2 q3 = *(const int2*)(p + 48);
  v8i b = {q0.x, q0.y, q1.x, q1.y, q2.x, q2.y, q3.x, q3.y};
  return b;
}

__global__ __launch_bounds__(256) void gemm1_swiglu(const int8_t* __restrict__ Aq,
                                                    const int8_t* __restrict__ Wq,
                                                    const float* __restrict__ inv_a,
                                                    const float* __restrict__ scal,
                                                    __hip_bfloat16* __restrict__ Hf) {
  __shared__ alignas(16) int8_t As[2][128 * 80];
  __shared__ alignas(16) int8_t Bv[2][128 * 80];
  __shared__ alignas(16) int8_t Bg[2][128 * 80];
  const int tid = threadIdx.x;
  const int wave = tid >> 5, lane = tid & 31;
  const int wm = wave & 3, wn = wave >> 2;
  const int hi = lane >> 4, li = lane & 15;
  const int m0 = blockIdx.x * 128;
  const int n0 = blockIdx.y * 128;

  const int sr = tid >> 2;
  const int sc = (tid & 3) << 4;

  v8i accv[2][4] = {};
  v8i accg[2][4] = {};

  auto stage = [&](int k0, int buf) {
    async_copy16(&As[buf][sr * 80 + sc],        Aq + (size_t)(m0 + sr) * D_MODEL + k0 + sc);
    async_copy16(&As[buf][(sr + 64) * 80 + sc], Aq + (size_t)(m0 + sr + 64) * D_MODEL + k0 + sc);
    async_copy16(&Bv[buf][sr * 80 + sc],        Wq + (size_t)(n0 + sr) * D_MODEL + k0 + sc);
    async_copy16(&Bv[buf][(sr + 64) * 80 + sc], Wq + (size_t)(n0 + sr + 64) * D_MODEL + k0 + sc);
    async_copy16(&Bg[buf][sr * 80 + sc],        Wq + (size_t)(HIDDEN + n0 + sr) * D_MODEL + k0 + sc);
    async_copy16(&Bg[buf][(sr + 64) * 80 + sc], Wq + (size_t)(HIDDEN + n0 + sr + 64) * D_MODEL + k0 + sc);
  };

  auto compute = [&](int buf) {
    v8i af[2];
#pragma unroll
    for (int t = 0; t < 2; ++t) af[t] = load_a_frag(As[buf], wm * 32 + t * 16 + li, hi);
#pragma unroll
    for (int tn = 0; tn < 4; ++tn) {
      v8i bv = load_b_frag(Bv[buf], wn * 64 + tn * 16 + li, hi);
      v8i bg = load_b_frag(Bg[buf], wn * 64 + tn * 16 + li, hi);
#pragma unroll
      for (int tm = 0; tm < 2; ++tm) {
        accv[tm][tn] = __builtin_amdgcn_wmma_i32_16x16x64_iu8(true, af[tm], true, bv,
                                                              accv[tm][tn], false, false);
        accg[tm][tn] = __builtin_amdgcn_wmma_i32_16x16x64_iu8(true, af[tm], true, bg,
                                                              accg[tm][tn], false, false);
      }
    }
  };

  stage(0, 0);
  wait_async();
  __syncthreads();

  for (int k0 = 0; k0 < D_MODEL; k0 += 128) {
    stage(k0 + 64, 1);
    compute(0);
    wait_async();
    __syncthreads();
    stage(k0 + 128, 0);
    compute(1);
    wait_async();
    __syncthreads();
  }

  const float sw = scal[0];
#pragma unroll
  for (int tm = 0; tm < 2; ++tm)
#pragma unroll
    for (int tn = 0; tn < 4; ++tn) {
      const int n = n0 + wn * 64 + tn * 16 + li;
#pragma unroll
      for (int r = 0; r < 8; ++r) {
        const int m = m0 + wm * 32 + tm * 16 + hi * 8 + r;
        const float d = inv_a[m] * sw;
        const float v = (float)accv[tm][tn][r] * d;
        const float g = (float)accg[tm][tn][r] * d;
        const float h = v * (g / (1.f + __expf(-g)));
        Hf[(size_t)m * HIDDEN + n] = __float2bfloat16(h);
      }
    }
}

__global__ __launch_bounds__(256) void gemm2_residual(const int8_t* __restrict__ Hq,
                                                      const int8_t* __restrict__ Wq,
                                                      const float* __restrict__ inv_h,
                                                      const float* __restrict__ scal,
                                                      const float* __restrict__ X,
                                                      float* __restrict__ Out) {
  __shared__ alignas(16) int8_t As[2][128 * 80];
  __shared__ alignas(16) int8_t Bs[2][128 * 80];
  const int tid = threadIdx.x;
  const int wave = tid >> 5, lane = tid & 31;
  const int wm = wave & 3, wn = wave >> 2;
  const int hi = lane >> 4, li = lane & 15;
  const int m0 = blockIdx.x * 128;
  const int n0 = blockIdx.y * 128;

  const int sr = tid >> 2;
  const int sc = (tid & 3) << 4;

  v8i acc[2][4] = {};

  auto stage = [&](int k0, int buf) {
    async_copy16(&As[buf][sr * 80 + sc],        Hq + (size_t)(m0 + sr) * HIDDEN + k0 + sc);
    async_copy16(&As[buf][(sr + 64) * 80 + sc], Hq + (size_t)(m0 + sr + 64) * HIDDEN + k0 + sc);
    async_copy16(&Bs[buf][sr * 80 + sc],        Wq + (size_t)(n0 + sr) * HIDDEN + k0 + sc);
    async_copy16(&Bs[buf][(sr + 64) * 80 + sc], Wq + (size_t)(n0 + sr + 64) * HIDDEN + k0 + sc);
  };

  auto compute = [&](int buf) {
    v8i af[2];
#pragma unroll
    for (int t = 0; t < 2; ++t) af[t] = load_a_frag(As[buf], wm * 32 + t * 16 + li, hi);
#pragma unroll
    for (int tn = 0; tn < 4; ++tn) {
      v8i bf = load_b_frag(Bs[buf], wn * 64 + tn * 16 + li, hi);
#pragma unroll
      for (int tm = 0; tm < 2; ++tm)
        acc[tm][tn] = __builtin_amdgcn_wmma_i32_16x16x64_iu8(true, af[tm], true, bf,
                                                             acc[tm][tn], false, false);
    }
  };

  stage(0, 0);
  wait_async();
  __syncthreads();

  for (int k0 = 0; k0 < HIDDEN; k0 += 128) {
    stage(k0 + 64, 1);
    compute(0);
    wait_async();
    __syncthreads();
    stage(k0 + 128, 0);
    compute(1);
    wait_async();
    __syncthreads();
  }

  const float sw = scal[1];
#pragma unroll
  for (int tm = 0; tm < 2; ++tm)
#pragma unroll
    for (int tn = 0; tn < 4; ++tn) {
      const int n = n0 + wn * 64 + tn * 16 + li;
#pragma unroll
      for (int r = 0; r < 8; ++r) {
        const int m = m0 + wm * 32 + tm * 16 + hi * 8 + r;
        const size_t idx = (size_t)m * D_MODEL + n;
        Out[idx] = X[idx] + (float)acc[tm][tn][r] * (inv_h[m] * sw);
      }
    }
}

extern "C" void kernel_launch(void* const* d_in, const int* in_sizes, int n_in,
                              void* d_out, int out_size, void* d_ws, size_t ws_size,
                              hipStream_t stream) {
  (void)in_sizes; (void)n_in; (void)out_size; (void)ws_size;
  const float* x      = (const float*)d_in[0];
  const float* norm_w = (const float*)d_in[1];
  const float* w_gv   = (const float*)d_in[2];
  const float* w_out  = (const float*)d_in[3];

  char* ws = (char*)d_ws;
  int8_t* aq    = (int8_t*)(ws + OFF_AQ);
  int8_t* wgvq  = (int8_t*)(ws + OFF_WGVQ);
  int8_t* woutq = (int8_t*)(ws + OFF_WOUTQ);
  int8_t* hq    = (int8_t*)(ws + OFF_HQ);
  __hip_bfloat16* hf = (__hip_bfloat16*)(ws + OFF_HF);
  float* inv_a = (float*)(ws + OFF_INVA);
  float* inv_h = (float*)(ws + OFF_INVH);
  float* part  = (float*)(ws + OFF_PART);
  float* scal  = (float*)(ws + OFF_SCAL);

  const size_t n_gv  = (size_t)2 * HIDDEN * D_MODEL;
  const size_t n_out = (size_t)D_MODEL * HIDDEN;

  absmean_partial<<<1024, 256, 0, stream>>>(w_gv, n_gv, part);
  finalize_scale<<<1, 256, 0, stream>>>(part, 1024, 1.f / (float)n_gv, scal + 0);
  quant_weight<<<4096, 256, 0, stream>>>(w_gv, scal + 0, wgvq, n_gv);
  absmean_partial<<<1024, 256, 0, stream>>>(w_out, n_out, part);
  finalize_scale<<<1, 256, 0, stream>>>(part, 1024, 1.f / (float)n_out, scal + 1);
  quant_weight<<<4096, 256, 0, stream>>>(w_out, scal + 1, woutq, n_out);

  rmsnorm_actquant<<<TOKENS, 256, 0, stream>>>(x, norm_w, aq, inv_a);
  gemm1_swiglu<<<dim3(TOKENS / 128, HIDDEN / 128), 256, 0, stream>>>(aq, wgvq, inv_a, scal, hf);
  actquant_row<<<TOKENS, 256, 0, stream>>>(hf, hq, inv_h);
  gemm2_residual<<<dim3(TOKENS / 128, D_MODEL / 128), 256, 0, stream>>>(hq, woutq, inv_h, scal,
                                                                        x, (float*)d_out);
}
